// HyMultiHeadAttention_75651553952309
// MI455X (gfx1250) — matrix operand order repaired and checked by reading, not compiled, not run on hardware
//
#include <hip/hip_runtime.h>
#include <hip/hip_bf16.h>

typedef __attribute__((ext_vector_type(16))) _Float16 v16h;
typedef __attribute__((ext_vector_type(8)))  _Float16 v8h;
typedef __attribute__((ext_vector_type(8)))  float    v8f;

#define D_MODEL 1024
#define NHEAD   16
#define DK      64
#define BATCH   4
#define SEQ     2048
#define BS      (BATCH*SEQ)   // 8192 tokens

// ---------------------------------------------------------------------------
// WMMA fragment helpers (wave32, 16x16x32 f16 -> f32)
//  A (16x32, 16-bit): lane m = lane%16.
//    lanes 0-15 : halves 0-15 -> K 0-15
//    lanes16-31 : halves 0-15 -> K 16-31
//  B (32x16, 16-bit): lane col n = lane%16; 16 consecutive K per lane,
//    lanes 0-15 -> K 0-15, lanes 16-31 -> K 16-31.
//  C/D (16x16 f32): vgpr r -> row m = r + 8*(lane/16), col n = lane%16.
// ---------------------------------------------------------------------------
__device__ __forceinline__ v16h load_a_frag(const _Float16* rowbase, int lane) {
  const int lo = (lane < 16) ? 0 : 16;
  v8h a0 = *(const v8h*)(rowbase + lo);
  v8h a1 = *(const v8h*)(rowbase + lo + 8);
  return __builtin_shufflevector(a0, a1, 0,1,2,3,4,5,6,7,8,9,10,11,12,13,14,15);
}

#define WMMA_F16(A, B, C) \
  __builtin_amdgcn_wmma_f32_16x16x32_f16(false, (A), false, (B), (short)0, (C), false, false)

// ---------------------------------------------------------------------------
// Prep: x -> f16, lambda = 2/(1 - ||x||^2) per token
// ---------------------------------------------------------------------------
__global__ __launch_bounds__(256)
void prep_x_kernel(const float* __restrict__ x, _Float16* __restrict__ xh,
                   float* __restrict__ lam) {
  __shared__ float red[256];
  const int tok = blockIdx.x, tid = threadIdx.x;
  const float* xr = x + (size_t)tok * D_MODEL;
  _Float16* xo = xh + (size_t)tok * D_MODEL;
  float ss = 0.f;
  for (int i = tid; i < D_MODEL; i += 256) {
    float v = xr[i];
    xo[i] = (_Float16)v;
    ss += v * v;
  }
  red[tid] = ss;
  __syncthreads();
  for (int s = 128; s > 0; s >>= 1) {
    if (tid < s) red[tid] += red[tid + s];
    __syncthreads();
  }
  if (tid == 0) lam[tok] = 2.0f / (1.0f - red[0]);
}

// ---------------------------------------------------------------------------
// Prep hyperbolic weights: column norms of z, cosh/sinh(2r),
// f16 transposed normalized weight zt[n][k] = z[k][n]/||z[:,n]||
// ---------------------------------------------------------------------------
__global__ __launch_bounds__(256)
void prep_whyp_kernel(const float* __restrict__ z, const float* __restrict__ r,
                      _Float16* __restrict__ zt, float* __restrict__ zn,
                      float* __restrict__ ch, float* __restrict__ sh) {
  const int n = blockIdx.x * 256 + threadIdx.x;
  float ss = 0.f;
  for (int k = 0; k < D_MODEL; ++k) {
    float v = z[(size_t)k * D_MODEL + n];
    ss += v * v;
  }
  float norm = fmaxf(sqrtf(ss), 1e-15f);
  float inv = 1.0f / norm;
  float tc = 2.0f * r[n];             // 2*sqrt(c)*r, c = 1
  float e = __expf(tc), ei = 1.0f / e;
  zn[n] = norm;
  ch[n] = 0.5f * (e + ei);
  sh[n] = 0.5f * (e - ei);
  for (int k = 0; k < D_MODEL; ++k)
    zt[(size_t)n * D_MODEL + k] = (_Float16)(z[(size_t)k * D_MODEL + n] * inv);
}

__global__ __launch_bounds__(256)
void prep_wv_kernel(const float* __restrict__ wv, _Float16* __restrict__ wvh) {
  size_t i = (size_t)blockIdx.x * 256 + threadIdx.x;
  wvh[i] = (_Float16)wv[i];
}

// ---------------------------------------------------------------------------
// Projection GEMM: 16 tokens x full D outputs per workgroup (8 waves x 128 N).
// MODE 0: hyperbolic epilogue (Q/K), writes f16 row-major (scaled by outscale).
// MODE 1: linear +bias (V), writes f16 TRANSPOSED vt[feature][token].
// ---------------------------------------------------------------------------
template <int MODE>
__global__ __launch_bounds__(256)
void proj_kernel(const _Float16* __restrict__ xh, const _Float16* __restrict__ wt,
                 const float* __restrict__ zn, const float* __restrict__ ch,
                 const float* __restrict__ sh, const float* __restrict__ lam,
                 const float* __restrict__ bias, _Float16* __restrict__ outp,
                 float outscale) {
  const int tid = threadIdx.x;
  const int lane = tid & 31, w = tid >> 5;
  const int lhalf = lane >> 4, l16 = lane & 15;
  const int tokbase = blockIdx.x * 16;
  const int n0 = w * 128;

  v8f acc[8];
#pragma unroll
  for (int t = 0; t < 8; ++t) acc[t] = v8f{};

  const _Float16* xrow = xh + (size_t)(tokbase + l16) * D_MODEL;
  for (int kk = 0; kk < D_MODEL / 32; ++kk) {
    v16h a = load_a_frag(xrow + kk * 32, lane);
    v16h bfr[8];
#pragma unroll
    for (int t = 0; t < 8; ++t) {
      const _Float16* bp =
          wt + (size_t)(n0 + t * 16 + l16) * D_MODEL + kk * 32 + 16 * lhalf;
      bfr[t] = *(const v16h*)bp;
    }
#pragma unroll
    for (int t = 0; t < 8; ++t) acc[t] = WMMA_F16(a, bfr[t], acc[t]);
  }

  if constexpr (MODE == 0) {
    __shared__ float wbuf[16 * 1024];   // 64 KB of CDNA5 LDS
    float lamr[8];
#pragma unroll
    for (int r = 0; r < 8; ++r) lamr[r] = lam[tokbase + r + 8 * lhalf];
#pragma unroll
    for (int t = 0; t < 8; ++t) {
      const int n = n0 + t * 16 + l16;
      const float chn = ch[n], shn = sh[n], znn = zn[n];
#pragma unroll
      for (int r = 0; r < 8; ++r) {
        float g = lamr[r] * acc[t][r] * chn - (lamr[r] - 1.0f) * shn;
        float as = copysignf(__logf(fabsf(g) + sqrtf(g * g + 1.0f)), g);
        float v = 2.0f * znn * as;                 // (2*zn/sqrt(c))*asinh(.)
        float wv = 0.5f * (__expf(v) - __expf(-v)); // sinh(v)
        wbuf[(r + 8 * lhalf) * 1024 + n] = wv;
      }
    }
    __syncthreads();
    // row-wise ball projection: q = w / (1 + sqrt(1 + ||w||^2))
    const int rr = tid >> 4, ii = tid & 15;
    const float* row = &wbuf[rr * 1024 + ii * 64];
    float ss = 0.f;
    for (int j = 0; j < 64; ++j) { float vv = row[j]; ss += vv * vv; }
    ss += __shfl_xor(ss, 1); ss += __shfl_xor(ss, 2);
    ss += __shfl_xor(ss, 4); ss += __shfl_xor(ss, 8);
    const float sc = outscale / (1.0f + sqrtf(1.0f + ss));
    _Float16* orow = outp + (size_t)(tokbase + rr) * D_MODEL + ii * 64;
    for (int j = 0; j < 64; ++j) orow[j] = (_Float16)(row[j] * sc);
  } else {
#pragma unroll
    for (int t = 0; t < 8; ++t) {
      const int n = n0 + t * 16 + l16;
      const float bn = bias[n];
#pragma unroll
      for (int r = 0; r < 8; ++r) {
        const int tok = tokbase + r + 8 * lhalf;
        outp[(size_t)n * BS + tok] = (_Float16)(acc[t][r] + bn);
      }
    }
  }
}

// ---------------------------------------------------------------------------
// Attention: grid (S/256, B*H), 8 waves, each wave owns 32 q-rows (2 M-blocks).
//
// Numerical note: Q,K live in the Poincare unit ball, so every 64-dim head
// slice has norm < 1 and |q.k|/sqrt(64) < 0.125 (scale pre-folded into Q).
// exp(s) is in [0.88, 1.13] -> the shift-free softmax P = exp(s), l = sum
// exp(s) is exactly correct and cannot overflow; row sums are accumulated on
// the matrix pipe as l += P @ ones.  K tiles are double-buffered in registers.
// ---------------------------------------------------------------------------
__global__ __launch_bounds__(256)
void attn_kernel(const _Float16* __restrict__ qh, const _Float16* __restrict__ kh,
                 const _Float16* __restrict__ vt, float* __restrict__ out) {
  __shared__ _Float16 plds[8][32 * 32];   // per-wave P relayout buffer (16 KB)
  const int tid = threadIdx.x;
  const int lane = tid & 31, w = tid >> 5;
  const int lhalf = lane >> 4, l16 = lane & 15;
  const int bh = blockIdx.y;
  const int b = bh >> 4, h = bh & 15;
  const int hf = h * DK;
  const int qtok = b * SEQ + blockIdx.x * 256 + w * 32;

  v16h aq[2][2];
#pragma unroll
  for (int m = 0; m < 2; ++m) {
    const _Float16* qrow = qh + (size_t)(qtok + m * 16 + l16) * D_MODEL + hf;
    aq[m][0] = load_a_frag(qrow, lane);
    aq[m][1] = load_a_frag(qrow + 32, lane);
  }

  v16h ones;
#pragma unroll
  for (int i = 0; i < 16; ++i) ones[i] = (_Float16)1.0f;

  v8f o[2][4];
  v8f la[2];
#pragma unroll
  for (int m = 0; m < 2; ++m) {
    la[m] = v8f{};
#pragma unroll
    for (int t = 0; t < 4; ++t) o[m][t] = v8f{};
  }

  // K tile register double-buffer
  v16h bk[2][2];   // [ntile][kstep]
  {
    const int kt0 = b * SEQ;
#pragma unroll
    for (int nt = 0; nt < 2; ++nt) {
      const _Float16* kr =
          kh + (size_t)(kt0 + nt * 16 + l16) * D_MODEL + hf + 16 * lhalf;
      bk[nt][0] = *(const v16h*)(kr);
      bk[nt][1] = *(const v16h*)(kr + 32);
    }
  }

  for (int jt = 0; jt < SEQ / 32; ++jt) {
    const int kt = b * SEQ + jt * 32;

    // V B-frags for this tile (independent of softmax -> issue early)
    const _Float16* vb = vt + (size_t)(hf + l16) * BS + kt + 16 * lhalf;
    v16h bv0 = *(const v16h*)(vb);
    v16h bv1 = *(const v16h*)(vb + (size_t)16 * BS);
    v16h bv2 = *(const v16h*)(vb + (size_t)32 * BS);
    v16h bv3 = *(const v16h*)(vb + (size_t)48 * BS);
    if (jt + 1 < SEQ / 32) {
      __builtin_prefetch(vb + 32, 0, 0);
      __builtin_prefetch(vb + (size_t)48 * BS + 32, 0, 0);
    }

    // scores (Q pre-scaled by 1/sqrt(Dk))
    v8f s00 = v8f{}, s01 = v8f{}, s10 = v8f{}, s11 = v8f{};
    s00 = WMMA_F16(aq[0][0], bk[0][0], s00);
    s00 = WMMA_F16(aq[0][1], bk[0][1], s00);
    s01 = WMMA_F16(aq[0][0], bk[1][0], s01);
    s01 = WMMA_F16(aq[0][1], bk[1][1], s01);
    s10 = WMMA_F16(aq[1][0], bk[0][0], s10);
    s10 = WMMA_F16(aq[1][1], bk[0][1], s10);
    s11 = WMMA_F16(aq[1][0], bk[1][0], s11);
    s11 = WMMA_F16(aq[1][1], bk[1][1], s11);

    // load NEXT K tile while this tile's softmax/PV proceeds
    if (jt + 1 < SEQ / 32) {
      const int ktn = kt + 32;
#pragma unroll
      for (int nt = 0; nt < 2; ++nt) {
        const _Float16* kr =
            kh + (size_t)(ktn + nt * 16 + l16) * D_MODEL + hf + 16 * lhalf;
        bk[nt][0] = *(const v16h*)(kr);
        bk[nt][1] = *(const v16h*)(kr + 32);
      }
    }

    // shift-free softmax numerator, relayout C->A via per-wave LDS
    _Float16* pl = &plds[w][0];
#pragma unroll
    for (int r = 0; r < 8; ++r) {
      const int row0 = (r + 8 * lhalf) * 32;
      const int row1 = (16 + r + 8 * lhalf) * 32;
      pl[row0 + l16]      = (_Float16)__expf(s00[r]);
      pl[row0 + 16 + l16] = (_Float16)__expf(s01[r]);
      pl[row1 + l16]      = (_Float16)__expf(s10[r]);
      pl[row1 + 16 + l16] = (_Float16)__expf(s11[r]);
    }
    asm volatile("s_wait_dscnt 0x0" ::: "memory");
    v16h pa0 = load_a_frag(pl + l16 * 32, lane);
    v16h pa1 = load_a_frag(pl + (16 + l16) * 32, lane);

    // O += P @ V ; l += P @ ones   (all on the matrix pipe)
    o[0][0] = WMMA_F16(pa0, bv0, o[0][0]);
    o[0][1] = WMMA_F16(pa0, bv1, o[0][1]);
    o[0][2] = WMMA_F16(pa0, bv2, o[0][2]);
    o[0][3] = WMMA_F16(pa0, bv3, o[0][3]);
    la[0]   = WMMA_F16(pa0, ones, la[0]);
    o[1][0] = WMMA_F16(pa1, bv0, o[1][0]);
    o[1][1] = WMMA_F16(pa1, bv1, o[1][1]);
    o[1][2] = WMMA_F16(pa1, bv2, o[1][2]);
    o[1][3] = WMMA_F16(pa1, bv3, o[1][3]);
    la[1]   = WMMA_F16(pa1, ones, la[1]);
  }

#pragma unroll
  for (int m = 0; m < 2; ++m) {
#pragma unroll
    for (int r = 0; r < 8; ++r) {
      const float inv = 1.0f / la[m][r];
      const int tok = qtok + m * 16 + r + 8 * lhalf;
      float* orow = out + (size_t)tok * D_MODEL + hf + l16;
      orow[0]  = o[m][0][r] * inv;
      orow[16] = o[m][1][r] * inv;
      orow[32] = o[m][2][r] * inv;
      orow[48] = o[m][3][r] * inv;
    }
  }
}

// ---------------------------------------------------------------------------
extern "C" void kernel_launch(void* const* d_in, const int* in_sizes, int n_in,
                              void* d_out, int out_size, void* d_ws, size_t ws_size,
                              hipStream_t stream) {
  const float* x  = (const float*)d_in[0];
  const float* zq = (const float*)d_in[1];
  const float* bq = (const float*)d_in[2];
  const float* zk = (const float*)d_in[3];
  const float* bk = (const float*)d_in[4];
  const float* wv = (const float*)d_in[5];
  const float* bv = (const float*)d_in[6];

  char* ws = (char*)d_ws;
  _Float16* xh  = (_Float16*)ws; ws += (size_t)BS * D_MODEL * sizeof(_Float16);
  _Float16* qh  = (_Float16*)ws; ws += (size_t)BS * D_MODEL * sizeof(_Float16);
  _Float16* kh  = (_Float16*)ws; ws += (size_t)BS * D_MODEL * sizeof(_Float16);
  _Float16* vt  = (_Float16*)ws; ws += (size_t)D_MODEL * BS * sizeof(_Float16);
  _Float16* ztq = (_Float16*)ws; ws += (size_t)D_MODEL * D_MODEL * sizeof(_Float16);
  _Float16* ztk = (_Float16*)ws; ws += (size_t)D_MODEL * D_MODEL * sizeof(_Float16);
  _Float16* wvh = (_Float16*)ws; ws += (size_t)D_MODEL * D_MODEL * sizeof(_Float16);
  float* lam = (float*)ws; ws += (size_t)BS * sizeof(float);
  float* znq = (float*)ws; ws += D_MODEL * sizeof(float);
  float* chq = (float*)ws; ws += D_MODEL * sizeof(float);
  float* shq = (float*)ws; ws += D_MODEL * sizeof(float);
  float* znk = (float*)ws; ws += D_MODEL * sizeof(float);
  float* chk = (float*)ws; ws += D_MODEL * sizeof(float);
  float* shk = (float*)ws; ws += D_MODEL * sizeof(float);

  prep_x_kernel<<<BS, 256, 0, stream>>>(x, xh, lam);
  prep_whyp_kernel<<<D_MODEL / 256, 256, 0, stream>>>(zq, bq, ztq, znq, chq, shq);
  prep_whyp_kernel<<<D_MODEL / 256, 256, 0, stream>>>(zk, bk, ztk, znk, chk, shk);
  prep_wv_kernel<<<(D_MODEL * D_MODEL) / 256, 256, 0, stream>>>(wv, wvh);

  // Q is pre-scaled by 1/sqrt(Dk) so attention needs no score scaling.
  proj_kernel<0><<<BS / 16, 256, 0, stream>>>(xh, ztq, znq, chq, shq, lam, nullptr, qh, 0.125f);
  proj_kernel<0><<<BS / 16, 256, 0, stream>>>(xh, ztk, znk, chk, shk, lam, nullptr, kh, 1.0f);
  proj_kernel<1><<<BS / 16, 256, 0, stream>>>(xh, wvh, nullptr, nullptr, nullptr, nullptr, bv, vt, 1.0f);

  attn_kernel<<<dim3(SEQ / 256, BATCH * NHEAD), 256, 0, stream>>>(qh, kh, vt, (float*)d_out);
}
